// SingleAttentionBlockRef_69106023793462
// MI455X (gfx1250) — compile-verified, matrix operand order checked by reading, not run on hardware
//
#include <hip/hip_runtime.h>
#include <cstddef>

// ---------------------------------------------------------------------------
// MI455X (gfx1250, wave32) implementation of the dual-stream attention block.
// All large matmuls use v_wmma_f32_16x16x32_bf16 (bf16 in, fp32 accumulate).
// LDS tile staging uses GLOBAL_LOAD_ASYNC_TO_LDS (ASYNCcnt) when the
// toolchain exposes the gfx1250 builtins, else falls back to load+ds_store.
// ---------------------------------------------------------------------------

typedef __bf16 bf16;
typedef __attribute__((ext_vector_type(16))) __bf16 v16bf;
typedef __attribute__((ext_vector_type(8)))  float  v8f;
typedef __attribute__((ext_vector_type(4)))  int    v4i;

struct __align__(16) U128 { unsigned int x[4]; };
struct __align__(16) F128 { float x[4]; };

#define AS1 __attribute__((address_space(1)))
#define AS3 __attribute__((address_space(3)))

#if defined(__has_builtin)
#  if __has_builtin(__builtin_amdgcn_global_load_async_to_lds_b128) && \
      __has_builtin(__builtin_amdgcn_s_wait_asynccnt)
#    define USE_ASYNC_LDS 1
#  endif
#endif
#ifndef USE_ASYNC_LDS
#  define USE_ASYNC_LDS 0
#endif

#define SEQ_T 2304
#define DIM_T 2048
#define FFN_T 8192
#define NHEAD 16
#define HDIM  128
#define CSEQ  1152      // h * 2w = 24 * 48 (combined seq per frame)
#define NBH   64        // 4 frames * 16 heads

// ------------------------------ helpers ------------------------------------

__device__ __forceinline__ float siluf(float x) { return x / (1.f + __expf(-x)); }

__device__ __forceinline__ float gelu_tanh(float x) {
    float x3 = x * x * x;
    return 0.5f * x * (1.f + tanhf(0.7978845608028654f * (x + 0.044715f * x3)));
}

// 16-byte global -> LDS copy; async (ASYNCcnt) on gfx1250 toolchains that
// expose the builtin, otherwise synchronous load + ds_store.
// Builtin signature (from hipcc diagnostic): (v4i AS1*, v4i AS3*, Imm, Imm).
__device__ __forceinline__ void cp_b128(const void* gsrc, void* ldst) {
#if USE_ASYNC_LDS
    __builtin_amdgcn_global_load_async_to_lds_b128(
        (AS1 v4i*)const_cast<void*>(gsrc), (AS3 v4i*)ldst, 0, 0);
#else
    *(U128*)ldst = *(const U128*)gsrc;
#endif
}

__device__ __forceinline__ void cp_wait() {
#if USE_ASYNC_LDS
    __builtin_amdgcn_s_wait_asynccnt(0);   // s_wait_asynccnt 0
#endif
}

// WMMA bf16 fragment loader. LDS tile is row-major [rows][ldk] bf16.
// A-matrix 16x32 layout (ISA 7.12.2): lane L holds row M = L%16;
// elements 0..7  -> K = base+0..7,  elements 8..15 -> K = base+16..23,
// where base = 8*(L/16). B-fragments are the transpose-dual: identical
// addressing applied to B^T stored row-major.
__device__ __forceinline__ v16bf load_frag_row(const bf16* base, int ldk,
                                               int row, int k0, int lane) {
    const bf16* p = base + row * ldk + k0 + ((lane >> 4) << 3);
    union { U128 q[2]; v16bf v; } u;
    u.q[0] = *(const U128*)(p);
    u.q[1] = *(const U128*)(p + 16);
    return u.v;
}

__device__ __forceinline__ v8f wmma_bf16(v16bf a, v16bf b, v8f c) {
    return __builtin_amdgcn_wmma_f32_16x16x32_bf16(
        /*neg_a=*/false, a, /*neg_b=*/false, b,
        /*c_mod=*/(short)0, c, /*reuse_a=*/false, /*reuse_b=*/false);
}

__device__ __forceinline__ float block_sum256(float v, float* sred) {
    const int lane = threadIdx.x & 31;
    const int wid  = threadIdx.x >> 5;
#pragma unroll
    for (int off = 16; off > 0; off >>= 1) v += __shfl_xor(v, off, 32);
    if (lane == 0) sred[wid] = v;
    __syncthreads();
    if (threadIdx.x == 0) {
        float s = 0.f;
#pragma unroll
        for (int i = 0; i < 8; i++) s += sred[i];
        sred[0] = s;
    }
    __syncthreads();
    float r = sred[0];
    __syncthreads();
    return r;
}

// ------------------------- tiny adaLN kernels ------------------------------

__global__ __launch_bounds__(256) void silu_kernel(const float* __restrict__ temb,
                                                   float* __restrict__ st) {
    int i = blockIdx.x * 256 + threadIdx.x;
    if (i < 512) st[i] = siluf(temb[i]);
}

// t[n] = sum_k st[k] * lw[k*6144+n] + lb[n]     (M=1 GEMM, N=6144, K=512)
__global__ __launch_bounds__(256) void adaln_kernel(const float* __restrict__ st,
                                                    const float* __restrict__ lw,
                                                    const float* __restrict__ lb,
                                                    float* __restrict__ t) {
    int n = blockIdx.x * 256 + threadIdx.x;
    float acc = lb[n];
    for (int k = 0; k < 512; k++) acc += st[k] * lw[(size_t)k * 6144 + n];
    t[n] = acc;
}

// ------------------- weight fp32 [K][N] -> bf16 [N][K] ---------------------

__global__ __launch_bounds__(256) void transpose_bf16_kernel(
    const float* __restrict__ src, bf16* __restrict__ dst, int K, int N) {
    __shared__ float tile[32][33];
    const int nb = blockIdx.x * 32, kb = blockIdx.y * 32;
    const int tx = threadIdx.x, ty = threadIdx.y;   // 32 x 8
#pragma unroll
    for (int i = 0; i < 32; i += 8)
        tile[ty + i][tx] = src[(size_t)(kb + ty + i) * N + nb + tx];
    __syncthreads();
#pragma unroll
    for (int i = 0; i < 32; i += 8)
        dst[(size_t)(nb + ty + i) * K + kb + tx] = (bf16)tile[tx][ty + i];
}

// --------------- LayerNorm + adaLN modulation -> bf16 ----------------------

__global__ __launch_bounds__(256) void ln_mod_kernel(
    const float* __restrict__ x, const float* __restrict__ t,
    const float* __restrict__ nw, const float* __restrict__ nb,
    bf16* __restrict__ out) {
    __shared__ float sred[8];
    const int row = blockIdx.x;
    const int tid = threadIdx.x;
    const float* xr = x + (size_t)row * DIM_T;
    float vals[8];
    float s = 0.f;
#pragma unroll
    for (int i = 0; i < 8; i++) { vals[i] = xr[tid + i * 256]; s += vals[i]; }
    float mu = block_sum256(s, sred) * (1.f / DIM_T);
    float vs = 0.f;
#pragma unroll
    for (int i = 0; i < 8; i++) { float d = vals[i] - mu; vs += d * d; }
    float var = block_sum256(vs, sred) * (1.f / DIM_T);
    float rs = rsqrtf(var + 1e-5f);
#pragma unroll
    for (int i = 0; i < 8; i++) {
        int c = tid + i * 256;
        float xn = (vals[i] - mu) * rs * nw[c] + nb[c];
        out[(size_t)row * DIM_T + c] = (bf16)(xn * (1.f + t[DIM_T + c]) + t[c]);
    }
}

// --------------------- bf16 WMMA GEMM (128x128x32) -------------------------
// C[M,N] = A[M,K] * Bt[N,K]^T ; ACT: 0=none 1=gelu-tanh
// OMODE: 0 -> outF = acc+bias ; 1 -> outB = bf16(act(acc+bias))
//        2 -> outF = resid + gate[n]*(acc+bias)

template <int ACT, int OMODE>
__global__ __launch_bounds__(256) void gemm_bf16_kernel(
    const bf16* __restrict__ A, const bf16* __restrict__ Bt,
    const float* __restrict__ bias, const float* __restrict__ resid,
    const float* __restrict__ gate, float* __restrict__ outF,
    bf16* __restrict__ outB, int M, int N, int K) {
    __shared__ __align__(16) bf16 As[128 * 32];
    __shared__ __align__(16) bf16 Bs[128 * 32];
    const int tid  = threadIdx.x;
    const int lane = tid & 31;
    const int wid  = tid >> 5;
    const int wm   = wid & 3;    // 4 waves along M (32 rows each)
    const int wn   = wid >> 2;   // 2 waves along N (64 cols each)
    const int m0   = blockIdx.y * 128;
    const int n0   = blockIdx.x * 128;

    const v8f vzero = {0.f, 0.f, 0.f, 0.f, 0.f, 0.f, 0.f, 0.f};
    v8f acc[2][4];
#pragma unroll
    for (int i = 0; i < 2; i++)
#pragma unroll
        for (int j = 0; j < 4; j++) acc[i][j] = vzero;

    const int lr = tid >> 2, lc = (tid & 3) << 3;   // this thread's slab element

    for (int k0 = 0; k0 < K; k0 += 32) {
        __syncthreads();
        for (int i = tid; i < 512; i += 256) {
            int r = i >> 2, c = (i & 3) << 3;
            cp_b128(&A[(size_t)(m0 + r) * K + k0 + c], &As[r * 32 + c]);
            cp_b128(&Bt[(size_t)(n0 + r) * K + k0 + c], &Bs[r * 32 + c]);
        }
        if (k0 + 32 < K) {   // hint the next K-slab into cache (global_prefetch_b8)
            __builtin_prefetch(&A[(size_t)(m0 + lr) * K + k0 + 32 + lc], 0, 0);
            __builtin_prefetch(&Bt[(size_t)(n0 + lr) * K + k0 + 32 + lc], 0, 0);
        }
        cp_wait();
        __syncthreads();
        v16bf af[2], bfv[4];
#pragma unroll
        for (int i = 0; i < 2; i++)
            af[i] = load_frag_row(As, 32, wm * 32 + i * 16 + (lane & 15), 0, lane);
#pragma unroll
        for (int j = 0; j < 4; j++)
            bfv[j] = load_frag_row(Bs, 32, wn * 64 + j * 16 + (lane & 15), 0, lane);
#pragma unroll
        for (int i = 0; i < 2; i++)
#pragma unroll
            for (int j = 0; j < 4; j++)
                acc[i][j] = wmma_bf16(af[i], bfv[j], acc[i][j]);
    }

    // epilogue: C-layout (ISA 7.12.2): vgpr r holds (m = r + 8*(lane/16), n = lane%16)
    union { v8f v; float f[8]; } u;
#pragma unroll
    for (int i = 0; i < 2; i++) {
#pragma unroll
        for (int j = 0; j < 4; j++) {
            u.v = acc[i][j];
            int col = n0 + wn * 64 + j * 16 + (lane & 15);
            int rowBase = m0 + wm * 32 + i * 16 + ((lane >> 4) << 3);
            float b = bias[col];
#pragma unroll
            for (int r = 0; r < 8; r++) {
                float v = u.f[r] + b;
                if (ACT == 1) v = gelu_tanh(v);
                size_t idx = (size_t)(rowBase + r) * N + col;
                if (OMODE == 0)      outF[idx] = v;
                else if (OMODE == 1) outB[idx] = (bf16)v;
                else                 outF[idx] = resid[idx] + gate[col] * v;
            }
        }
    }
}

// --------- RMSNorm + RoPE + scatter into combined attention layout ---------
// Qc/Kc: [bh=4*16][CSEQ][128] bf16 ; Vt: [bh][128][CSEQ] bf16 (pre-transposed)

__global__ __launch_bounds__(256) void qkv_finalize_kernel(
    const float* __restrict__ qf, const float* __restrict__ kf,
    const float* __restrict__ vf, const float* __restrict__ nqw,
    const float* __restrict__ nkw, const float* __restrict__ cosb,
    const float* __restrict__ sinb, bf16* __restrict__ Qc,
    bf16* __restrict__ Kc, bf16* __restrict__ Vt, int isRef) {
    __shared__ float sred[8];
    const int s   = blockIdx.x;       // 0..2303
    const int tid = threadIdx.x;
    const int bfi = s / 576, rem = s % 576;
    const int y = rem / 24, x = rem % 24;
    const int crow = y * 48 + x + (isRef ? 24 : 0);
    const int c0 = tid * 8;
    const int head = c0 >> 7, d0 = c0 & 127;
    const int bh = bfi * NHEAD + head;

    const float* qr = qf + (size_t)s * DIM_T;
    const float* kr = kf + (size_t)s * DIM_T;
    const float* vr = vf + (size_t)s * DIM_T;
    float qv[8], kv[8], vv[8];
    {
        F128 a0 = *(const F128*)(qr + c0), a1 = *(const F128*)(qr + c0 + 4);
        F128 b0 = *(const F128*)(kr + c0), b1 = *(const F128*)(kr + c0 + 4);
        F128 c0v = *(const F128*)(vr + c0), c1v = *(const F128*)(vr + c0 + 4);
#pragma unroll
        for (int i = 0; i < 4; i++) {
            qv[i] = a0.x[i]; qv[i + 4] = a1.x[i];
            kv[i] = b0.x[i]; kv[i + 4] = b1.x[i];
            vv[i] = c0v.x[i]; vv[i + 4] = c1v.x[i];
        }
    }
    float sq = 0.f, sk = 0.f;
#pragma unroll
    for (int i = 0; i < 8; i++) { sq += qv[i] * qv[i]; sk += kv[i] * kv[i]; }
    float qrs = rsqrtf(block_sum256(sq, sred) * (1.f / DIM_T) + 1e-6f);
    float krs = rsqrtf(block_sum256(sk, sred) * (1.f / DIM_T) + 1e-6f);

    float qn[8], kn[8];
#pragma unroll
    for (int i = 0; i < 8; i++) {
        qn[i] = qv[i] * qrs * nqw[c0 + i];
        kn[i] = kv[i] * krs * nkw[c0 + i];
    }
    const size_t qb = ((size_t)bh * CSEQ + crow) * HDIM + d0;
#pragma unroll
    for (int p = 0; p < 4; p++) {        // RoPE: even=x1*c - x2*s ; odd=x1*s + x2*c
        int d = d0 + 2 * p;
        float cc = cosb[(size_t)s * HDIM + d];
        float ss = sinb[(size_t)s * HDIM + d + 1];
        Qc[qb + 2 * p]     = (bf16)(qn[2 * p] * cc - qn[2 * p + 1] * ss);
        Qc[qb + 2 * p + 1] = (bf16)(qn[2 * p] * ss + qn[2 * p + 1] * cc);
        Kc[qb + 2 * p]     = (bf16)(kn[2 * p] * cc - kn[2 * p + 1] * ss);
        Kc[qb + 2 * p + 1] = (bf16)(kn[2 * p] * ss + kn[2 * p + 1] * cc);
    }
#pragma unroll
    for (int i = 0; i < 8; i++)
        Vt[((size_t)bh * HDIM + d0 + i) * CSEQ + crow] = (bf16)vv[i];
}

// ------------------------ flash attention (WMMA) ---------------------------
// grid (18 q-tiles, 64 bh), 128 threads = 4 waves; each wave owns 16 q rows.

__global__ __launch_bounds__(128) void flash_attn_kernel(
    const bf16* __restrict__ Qc, const bf16* __restrict__ Kc,
    const bf16* __restrict__ Vt, bf16* __restrict__ outH,
    bf16* __restrict__ outR) {
    __shared__ __align__(16) bf16 Qs[64 * 128];
    __shared__ __align__(16) bf16 Ks[64 * 128];
    __shared__ __align__(16) bf16 Vs[128 * 64];   // [d][key] = V^T tile
    __shared__ __align__(16) bf16 Ps[4 * 16 * 64];
    const int tid = threadIdx.x, lane = tid & 31, w = tid >> 5;
    const int qt = blockIdx.x, bh = blockIdx.y;
    const size_t qkb = (size_t)bh * CSEQ * HDIM;

    for (int i = tid; i < 1024; i += 128) {
        int r = i >> 4, c = (i & 15) << 3;
        cp_b128(&Qc[qkb + (size_t)(qt * 64 + r) * HDIM + c], &Qs[r * 128 + c]);
    }
    cp_wait();
    __syncthreads();
    v16bf qa[4];
#pragma unroll
    for (int ks = 0; ks < 4; ks++)
        qa[ks] = load_frag_row(Qs, 128, w * 16 + (lane & 15), ks * 32, lane);

    float mrow[8], lrow[8];
#pragma unroll
    for (int r = 0; r < 8; r++) { mrow[r] = -1e30f; lrow[r] = 0.f; }
    const v8f vzero = {0.f, 0.f, 0.f, 0.f, 0.f, 0.f, 0.f, 0.f};
    v8f oacc[8];
#pragma unroll
    for (int j = 0; j < 8; j++) oacc[j] = vzero;
    const float scale = 0.08838834764831845f;    // 128^-0.5

    for (int kt = 0; kt < CSEQ / 64; kt++) {
        __syncthreads();
        for (int i = tid; i < 1024; i += 128) {
            int r = i >> 4, c = (i & 15) << 3;
            cp_b128(&Kc[qkb + (size_t)(kt * 64 + r) * HDIM + c], &Ks[r * 128 + c]);
        }
        for (int i = tid; i < 1024; i += 128) {
            int r = i >> 3, c = (i & 7) << 3;
            cp_b128(&Vt[((size_t)bh * HDIM + r) * CSEQ + kt * 64 + c], &Vs[r * 64 + c]);
        }
        cp_wait();
        __syncthreads();

        v8f sac[4];
#pragma unroll
        for (int nt = 0; nt < 4; nt++) {
            sac[nt] = vzero;
#pragma unroll
            for (int ks = 0; ks < 4; ks++) {
                v16bf b = load_frag_row(Ks, 128, nt * 16 + (lane & 15), ks * 32, lane);
                sac[nt] = wmma_bf16(qa[ks], b, sac[nt]);
            }
        }
        union { v8f v; float f[8]; } su[4];
#pragma unroll
        for (int nt = 0; nt < 4; nt++) su[nt].v = sac[nt];
#pragma unroll
        for (int r = 0; r < 8; r++) {
            float mx = -1e30f;
#pragma unroll
            for (int nt = 0; nt < 4; nt++) {
                float s = su[nt].f[r] * scale;
                su[nt].f[r] = s;
                mx = fmaxf(mx, s);
            }
#pragma unroll
            for (int off = 8; off > 0; off >>= 1)
                mx = fmaxf(mx, __shfl_xor(mx, off, 32));   // within 16-lane half
            float mnew = fmaxf(mrow[r], mx);
            float rsum = 0.f;
            int lr = r + ((lane >> 4) << 3);
#pragma unroll
            for (int nt = 0; nt < 4; nt++) {
                float p = __expf(su[nt].f[r] - mnew);
                rsum += p;
                Ps[(w * 16 + lr) * 64 + nt * 16 + (lane & 15)] = (bf16)p;
            }
#pragma unroll
            for (int off = 8; off > 0; off >>= 1) rsum += __shfl_xor(rsum, off, 32);
            float corr = __expf(mrow[r] - mnew);
            lrow[r] = lrow[r] * corr + rsum;
            mrow[r] = mnew;
#pragma unroll
            for (int j = 0; j < 8; j++) oacc[j][r] *= corr;
        }
        __syncthreads();
#pragma unroll
        for (int ks = 0; ks < 2; ks++) {
            v16bf pa = load_frag_row(Ps + (w << 10), 64, (lane & 15), ks * 32, lane);
#pragma unroll
            for (int j = 0; j < 8; j++) {
                v16bf b = load_frag_row(Vs, 64, j * 16 + (lane & 15), ks * 32, lane);
                oacc[j] = wmma_bf16(pa, b, oacc[j]);
            }
        }
    }

    // scatter combined rows back into the two streams
    const int bfi = bh >> 4, head = bh & 15;
    union { v8f v; float f[8]; } uo;
#pragma unroll
    for (int j = 0; j < 8; j++) {
        uo.v = oacc[j];
#pragma unroll
        for (int r = 0; r < 8; r++) {
            int m = r + ((lane >> 4) << 3);
            int grow = qt * 64 + w * 16 + m;
            int d = j * 16 + (lane & 15);
            float val = uo.f[r] / lrow[r];
            int y = grow / 48, x = grow % 48;
            int sidx = bfi * 576 + y * 24 + (x % 24);
            bf16* dst = (x < 24) ? outH : outR;
            dst[(size_t)sidx * DIM_T + head * HDIM + d] = (bf16)val;
        }
    }
}

// ------------------------------ launcher -----------------------------------

extern "C" void kernel_launch(void* const* d_in, const int* in_sizes, int n_in,
                              void* d_out, int out_size, void* d_ws, size_t ws_size,
                              hipStream_t stream) {
    const float* hidden = (const float*)d_in[0];
    const float* ref    = (const float*)d_in[1];
    const float* temb   = (const float*)d_in[2];
    const float* cos_h  = (const float*)d_in[3];
    const float* sin_h  = (const float*)d_in[4];
    const float* cos_r  = (const float*)d_in[5];
    const float* sin_r  = (const float*)d_in[6];
    const float* ln1_lw = (const float*)d_in[7];
    const float* ln1_lb = (const float*)d_in[8];
    const float* ln1_nw = (const float*)d_in[9];
    const float* ln1_nb = (const float*)d_in[10];
    const float* ln2_lw = (const float*)d_in[11];
    const float* ln2_lb = (const float*)d_in[12];
    const float* ln2_nw = (const float*)d_in[13];
    const float* ln2_nb = (const float*)d_in[14];
    const float* wq = (const float*)d_in[15]; const float* bq = (const float*)d_in[16];
    const float* wk = (const float*)d_in[17]; const float* bk = (const float*)d_in[18];
    const float* wv = (const float*)d_in[19]; const float* bv = (const float*)d_in[20];
    const float* nq_w = (const float*)d_in[21];
    const float* nk_w = (const float*)d_in[22];
    const float* wo = (const float*)d_in[23]; const float* bo = (const float*)d_in[24];
    const float* w1 = (const float*)d_in[25]; const float* b1 = (const float*)d_in[26];
    const float* w2 = (const float*)d_in[27]; const float* b2 = (const float*)d_in[28];

    float* outHid = (float*)d_out;
    float* outRef = outHid + (size_t)SEQ_T * DIM_T;

    char* p = (char*)d_ws;
    auto alloc = [&](size_t bytes) -> char* {
        char* r = p;
        p += (bytes + 255) & ~(size_t)255;
        return r;
    };
    float* st  = (float*)alloc(512 * sizeof(float));
    float* t1  = (float*)alloc(6144 * sizeof(float));
    float* t2  = (float*)alloc(6144 * sizeof(float));
    bf16* wq_t = (bf16*)alloc((size_t)DIM_T * DIM_T * 2);
    bf16* wk_t = (bf16*)alloc((size_t)DIM_T * DIM_T * 2);
    bf16* wv_t = (bf16*)alloc((size_t)DIM_T * DIM_T * 2);
    bf16* wo_t = (bf16*)alloc((size_t)DIM_T * DIM_T * 2);
    bf16* w1_t = (bf16*)alloc((size_t)DIM_T * FFN_T * 2);   // [8192][2048]
    bf16* w2_t = (bf16*)alloc((size_t)FFN_T * DIM_T * 2);   // [2048][8192]
    bf16* nx   = (bf16*)alloc((size_t)SEQ_T * DIM_T * 2);
    float* qf  = (float*)alloc((size_t)SEQ_T * DIM_T * 4);
    float* kf  = (float*)alloc((size_t)SEQ_T * DIM_T * 4);
    float* vf  = (float*)alloc((size_t)SEQ_T * DIM_T * 4);
    bf16* Qc   = (bf16*)alloc((size_t)NBH * CSEQ * HDIM * 2);
    bf16* Kc   = (bf16*)alloc((size_t)NBH * CSEQ * HDIM * 2);
    bf16* Vt   = (bf16*)alloc((size_t)NBH * CSEQ * HDIM * 2);
    bf16* aoutH = (bf16*)alloc((size_t)SEQ_T * DIM_T * 2);
    bf16* aoutR = (bf16*)alloc((size_t)SEQ_T * DIM_T * 2);
    // aliases over dead buffers (sequential reuse, stream-ordered):
    float* hid1 = qf;              // fp32 hidden after attn residual
    float* ref1 = kf;              // fp32 ref    after attn residual
    bf16*  gbuf = Qc;              // ffn intermediate spans Qc..Kc (sizes match)

    // ---- adaLN modulation (depends only on temb; shared by both streams) --
    silu_kernel<<<2, 256, 0, stream>>>(temb, st);
    adaln_kernel<<<24, 256, 0, stream>>>(st, ln1_lw, ln1_lb, t1);
    adaln_kernel<<<24, 256, 0, stream>>>(st, ln2_lw, ln2_lb, t2);

    // ---- weight convert + transpose to bf16 [N][K] ------------------------
    dim3 tb(32, 8);
    transpose_bf16_kernel<<<dim3(DIM_T / 32, DIM_T / 32), tb, 0, stream>>>(wq, wq_t, DIM_T, DIM_T);
    transpose_bf16_kernel<<<dim3(DIM_T / 32, DIM_T / 32), tb, 0, stream>>>(wk, wk_t, DIM_T, DIM_T);
    transpose_bf16_kernel<<<dim3(DIM_T / 32, DIM_T / 32), tb, 0, stream>>>(wv, wv_t, DIM_T, DIM_T);
    transpose_bf16_kernel<<<dim3(DIM_T / 32, DIM_T / 32), tb, 0, stream>>>(wo, wo_t, DIM_T, DIM_T);
    transpose_bf16_kernel<<<dim3(FFN_T / 32, DIM_T / 32), tb, 0, stream>>>(w1, w1_t, DIM_T, FFN_T);
    transpose_bf16_kernel<<<dim3(DIM_T / 32, FFN_T / 32), tb, 0, stream>>>(w2, w2_t, FFN_T, DIM_T);

    dim3 gN(DIM_T / 128, SEQ_T / 128);   // (16, 18)
    dim3 gF(FFN_T / 128, SEQ_T / 128);   // (64, 18)

    // ---- stream H: ln1 -> qkv -> finalize ---------------------------------
    ln_mod_kernel<<<SEQ_T, 256, 0, stream>>>(hidden, t1, ln1_nw, ln1_nb, nx);
    gemm_bf16_kernel<0, 0><<<gN, 256, 0, stream>>>(nx, wq_t, bq, nullptr, nullptr, qf, nullptr, SEQ_T, DIM_T, DIM_T);
    gemm_bf16_kernel<0, 0><<<gN, 256, 0, stream>>>(nx, wk_t, bk, nullptr, nullptr, kf, nullptr, SEQ_T, DIM_T, DIM_T);
    gemm_bf16_kernel<0, 0><<<gN, 256, 0, stream>>>(nx, wv_t, bv, nullptr, nullptr, vf, nullptr, SEQ_T, DIM_T, DIM_T);
    qkv_finalize_kernel<<<SEQ_T, 256, 0, stream>>>(qf, kf, vf, nq_w, nk_w, cos_h, sin_h, Qc, Kc, Vt, 0);

    // ---- stream R ---------------------------------------------------------
    ln_mod_kernel<<<SEQ_T, 256, 0, stream>>>(ref, t1, ln1_nw, ln1_nb, nx);
    gemm_bf16_kernel<0, 0><<<gN, 256, 0, stream>>>(nx, wq_t, bq, nullptr, nullptr, qf, nullptr, SEQ_T, DIM_T, DIM_T);
    gemm_bf16_kernel<0, 0><<<gN, 256, 0, stream>>>(nx, wk_t, bk, nullptr, nullptr, kf, nullptr, SEQ_T, DIM_T, DIM_T);
    gemm_bf16_kernel<0, 0><<<gN, 256, 0, stream>>>(nx, wv_t, bv, nullptr, nullptr, vf, nullptr, SEQ_T, DIM_T, DIM_T);
    qkv_finalize_kernel<<<SEQ_T, 256, 0, stream>>>(qf, kf, vf, nq_w, nk_w, cos_r, sin_r, Qc, Kc, Vt, 1);

    // ---- joint attention over combined sequence ---------------------------
    flash_attn_kernel<<<dim3(CSEQ / 64, NBH), 128, 0, stream>>>(Qc, Kc, Vt, aoutH, aoutR);

    // ---- output projection + gated residual -------------------------------
    gemm_bf16_kernel<0, 2><<<gN, 256, 0, stream>>>(aoutH, wo_t, bo, hidden, t1 + 2 * DIM_T, hid1, nullptr, SEQ_T, DIM_T, DIM_T);
    gemm_bf16_kernel<0, 2><<<gN, 256, 0, stream>>>(aoutR, wo_t, bo, ref,    t1 + 2 * DIM_T, ref1, nullptr, SEQ_T, DIM_T, DIM_T);

    // ---- FFN (H): ln2 -> gelu(gemm1) -> gemm2 + gated residual ------------
    ln_mod_kernel<<<SEQ_T, 256, 0, stream>>>(hid1, t2, ln2_nw, ln2_nb, nx);
    gemm_bf16_kernel<1, 1><<<gF, 256, 0, stream>>>(nx, w1_t, b1, nullptr, nullptr, nullptr, gbuf, SEQ_T, FFN_T, DIM_T);
    gemm_bf16_kernel<0, 2><<<gN, 256, 0, stream>>>(gbuf, w2_t, b2, hid1, t2 + 2 * DIM_T, outHid, nullptr, SEQ_T, DIM_T, FFN_T);

    // ---- FFN (R) ----------------------------------------------------------
    ln_mod_kernel<<<SEQ_T, 256, 0, stream>>>(ref1, t2, ln2_nw, ln2_nb, nx);
    gemm_bf16_kernel<1, 1><<<gF, 256, 0, stream>>>(nx, w1_t, b1, nullptr, nullptr, nullptr, gbuf, SEQ_T, FFN_T, DIM_T);
    gemm_bf16_kernel<0, 2><<<gN, 256, 0, stream>>>(gbuf, w2_t, b2, ref1, t2 + 2 * DIM_T, outRef, nullptr, SEQ_T, DIM_T, FFN_T);

    (void)in_sizes; (void)n_in; (void)out_size; (void)ws_size;
}
